// NAOMIImputerAlpha_90640989815412
// MI455X (gfx1250) — matrix operand order repaired and checked by reading, not compiled, not run on hardware
//
#include <hip/hip_runtime.h>
#include <math.h>

typedef __attribute__((ext_vector_type(16))) _Float16 v16h;
typedef __attribute__((ext_vector_type(8)))  float    v8f;
typedef _Float16 half_t;

#define T_DIM   200
#define BSZ     64
#define P_DIM   22
#define RNN     256
#define G3      768                    // 3*RNN gates
#define DECD    128
#define NFWD    (P_DIM * T_DIM)        // 4400
#define NBWD    (P_DIM * (T_DIM - 1)) // 4378
#define STEP_H  (BSZ * RNN)            // 16384 elems per trace step

__device__ __forceinline__ float sigmoidf_(float x) {
  return 1.0f / (1.0f + __expf(-x));
}

// ---- WMMA fragment loaders (wave32, contiguous k = 16*half + element in both A and B) ----------------
// A 16x32 f16 (MxK), source row-major with leading dim lda.
__device__ __forceinline__ v16h load_a_frag(const half_t* A, int lda, int m0, int k0) {
  int lane = threadIdx.x & 31;
  int hh = lane >> 4;
  int m  = lane & 15;
  const half_t* row = A + (size_t)(m0 + m) * lda + k0;
  v16h out;
#pragma unroll
  for (int v = 0; v < 8; ++v) {
    int kb = hh * 16 + v * 2; // half*16 + v*2
    out[2 * v]     = row[kb];
    out[2 * v + 1] = row[kb + 1];
  }
  return out;
}

// B 32x16 f16 (KxN) where logical B[k][n] = W[n*K + k] (W row-major [N x K]).
__device__ __forceinline__ v16h load_b_fragT(const half_t* W, int K, int n0, int k0) {
  int lane = threadIdx.x & 31;
  int hh = lane >> 4;
  int n  = lane & 15;
  const half_t* col = W + (size_t)(n0 + n) * K + k0 + hh * 16;
  v16h out;
#pragma unroll
  for (int v = 0; v < 8; ++v) {
    out[2 * v]     = col[2 * v];
    out[2 * v + 1] = col[2 * v + 1];
  }
  return out;
}

// ---- block-wide gate GEMM: G(64x768) = A(64x256) @ W(768x256)^T + bias --
// 16 waves (512 threads); wave w owns N columns [w*48, w*48+48).
__device__ __forceinline__ void gemm_gate(const half_t* A, const half_t* W,
                                          const float* bias, half_t* G) {
  int wave = threadIdx.x >> 5;
  int lane = threadIdx.x & 31;
  int hh = lane >> 4, nl = lane & 15;
  int n0 = wave * 48;
#pragma unroll
  for (int mi = 0; mi < 4; ++mi) {
    v8f c0, c1, c2;
#pragma unroll
    for (int r = 0; r < 8; ++r) { c0[r] = 0.f; c1[r] = 0.f; c2[r] = 0.f; }
#pragma unroll
    for (int k0 = 0; k0 < RNN; k0 += 32) {
      v16h a  = load_a_frag(A, RNN, mi * 16, k0);
      v16h b0 = load_b_fragT(W, RNN, n0,      k0);
      v16h b1 = load_b_fragT(W, RNN, n0 + 16, k0);
      v16h b2 = load_b_fragT(W, RNN, n0 + 32, k0);
      c0 = __builtin_amdgcn_wmma_f32_16x16x32_f16(false, a, false, b0, (short)0, c0, false, false);
      c1 = __builtin_amdgcn_wmma_f32_16x16x32_f16(false, a, false, b1, (short)0, c1, false, false);
      c2 = __builtin_amdgcn_wmma_f32_16x16x32_f16(false, a, false, b2, (short)0, c2, false, false);
    }
#pragma unroll
    for (int r = 0; r < 8; ++r) {
      int m = mi * 16 + r + 8 * hh;
      int na = n0 + nl, nb = n0 + 16 + nl, nc = n0 + 32 + nl;
      G[m * G3 + na] = (half_t)(c0[r] + bias[na]);
      G[m * G3 + nb] = (half_t)(c1[r] + bias[nb]);
      G[m * G3 + nc] = (half_t)(c2[r] + bias[nc]);
    }
  }
}

// ---- persistent sequential GRU scan (block 0 = fwd, block 1 = bwd) ------
struct ScanArgs {
  const float* data; const float* gt;
  const float* f_wih0; const float* f_bih0; const float* f_bhh0;
  const float* f_bih1; const float* f_bhh1;
  const half_t* f_whh0; const half_t* f_whh1; const half_t* f_wih1;
  const float* b_wih0; const float* b_bih0; const float* b_bhh0;
  const float* b_bih1; const float* b_bhh1;
  const half_t* bw_whh0; const half_t* bw_whh1; const half_t* bw_wih1;
  half_t* hf_trace; half_t* hb_trace;
};

__global__ void __launch_bounds__(512) scan_kernel(ScanArgs A) {
  extern __shared__ char smem[];
  half_t* h0 = (half_t*)smem;              // 64x256 f16
  half_t* h1 = h0 + STEP_H;                // 64x256 f16
  half_t* GI = h1 + STEP_H;                // 64x768 f16
  half_t* GH = GI + BSZ * G3;              // 64x768 f16   (total 256 KB)

  const bool fwd = (blockIdx.x == 0);
  const int nsteps = fwd ? NFWD : NBWD;
  const int din = fwd ? 2 : 3;
  const float* wih0 = fwd ? A.f_wih0 : A.b_wih0;
  const float* bih0 = fwd ? A.f_bih0 : A.b_bih0;
  const float* bhh0 = fwd ? A.f_bhh0 : A.b_bhh0;
  const float* bih1 = fwd ? A.f_bih1 : A.b_bih1;
  const float* bhh1 = fwd ? A.f_bhh1 : A.b_bhh1;
  const half_t* whh0 = fwd ? A.f_whh0 : A.bw_whh0;
  const half_t* whh1 = fwd ? A.f_whh1 : A.bw_whh1;
  const half_t* wih1 = fwd ? A.f_wih1 : A.bw_wih1;
  half_t* trace = fwd ? A.hf_trace : A.hb_trace;

  for (int i = threadIdx.x; i < STEP_H; i += 512) {
    h0[i] = (half_t)0.f; h1[i] = (half_t)0.f;
  }
  __syncthreads();

  for (int step = 0; step < nsteps; ++step) {
    // layer 0 hidden-side gates: GH = h0 @ whh0^T + bhh0
    gemm_gate(h0, whh0, bhh0, GH);
    __syncthreads();

    // layer 0 input-side gates are tiny (din = 2 or 3): fold into elementwise.
    const float* xbase;
    int xstride;
    if (fwd) {
      int p = step / T_DIM, t = step % T_DIM;
      xbase = A.gt + ((size_t)t * BSZ * P_DIM + p) * 2;   // gt[t][b][p][:]
      xstride = P_DIM * 2;
    } else {
      int p = step / (T_DIM - 1), q = step % (T_DIM - 1);
      xbase = A.data + ((size_t)(T_DIM - 1 - q) * BSZ * P_DIM + p) * 3; // data[T-1-q][b][p][:]
      xstride = P_DIM * 3;
    }
    for (int idx = threadIdx.x; idx < STEP_H; idx += 512) {
      int b = idx >> 8, u = idx & 255;
      const float* x = xbase + b * xstride;
      float ir = bih0[u], iz = bih0[u + RNN], in = bih0[u + 2 * RNN];
      for (int c = 0; c < din; ++c) {
        float xv = x[c];
        ir += xv * wih0[u * din + c];
        iz += xv * wih0[(u + RNN) * din + c];
        in += xv * wih0[(u + 2 * RNN) * din + c];
      }
      float hr = (float)GH[b * G3 + u];
      float hz = (float)GH[b * G3 + u + RNN];
      float hn = (float)GH[b * G3 + u + 2 * RNN];
      float hp = (float)h0[idx];
      float r = sigmoidf_(ir + hr);
      float z = sigmoidf_(iz + hz);
      float n = tanhf(in + r * hn);
      h0[idx] = (half_t)((1.f - z) * n + z * hp);
    }
    __syncthreads();

    // layer 1: GI = h0new @ wih1^T + bih1 ; GH = h1 @ whh1^T + bhh1
    gemm_gate(h0, wih1, bih1, GI);
    gemm_gate(h1, whh1, bhh1, GH);
    __syncthreads();

    // scan emits h[-1] AFTER update (fwd) or BEFORE update (bwd -> shift by 1)
    bool wr = fwd || (step + 1 < nsteps);
    size_t tout = fwd ? (size_t)step : (size_t)(step + 1);
    half_t* tr = trace + tout * STEP_H;
    for (int idx = threadIdx.x; idx < STEP_H; idx += 512) {
      int b = idx >> 8, u = idx & 255;
      float ir = (float)GI[b * G3 + u];
      float iz = (float)GI[b * G3 + u + RNN];
      float in = (float)GI[b * G3 + u + 2 * RNN];
      float hr = (float)GH[b * G3 + u];
      float hz = (float)GH[b * G3 + u + RNN];
      float hn = (float)GH[b * G3 + u + 2 * RNN];
      float hp = (float)h1[idx];
      float r = sigmoidf_(ir + hr);
      float z = sigmoidf_(iz + hz);
      float n = tanhf(in + r * hn);
      half_t hv = (half_t)((1.f - z) * n + z * hp);
      h1[idx] = hv;
      if (wr) tr[idx] = hv;
    }
    __syncthreads();
  }
}

// ---- decoder + loss reduction ------------------------------------------
struct DecArgs {
  const float* gt;
  const half_t* hf; const half_t* hb;
  const half_t* decw[4];
  const float* decb[4];
  const float* meanw[4];
  const float* meanb[4];
  float* acc;
};

__global__ void __launch_bounds__(256) dec_kernel(DecArgs A) {
  __shared__ float dec[BSZ * DECD];   // 32 KB
  __shared__ float redl[BSZ];
  __shared__ float redp[BSZ];

  const int svals[4] = {1, 2, 4, 8};
  const int nts[4]   = {199, 197, 193, 185};   // T - 2s + 1
  int rem = blockIdx.x;
  int si = 0;
  while (si < 3 && rem >= P_DIM * nts[si]) { rem -= P_DIM * nts[si]; ++si; }
  const int s = svals[si], nt = nts[si];
  const int p = rem / nt, t = rem % nt;

  const half_t* Af = A.hf + (size_t)(p * T_DIM + t) * STEP_H;
  const int bstep = p * (T_DIM - 1) + (T_DIM - (t + 2 * s));
  const half_t* Ab = A.hb + (size_t)bstep * STEP_H;
  const half_t* W = A.decw[si];
  const float* db = A.decb[si];

  // GEMM: dec(64x128) = [Af | Ab](64x512) @ decw(128x512)^T ; 8 waves, 1 n-tile each
  int wave = threadIdx.x >> 5;
  int lane = threadIdx.x & 31;
  int hh = lane >> 4, nl = lane & 15;
  int n0 = wave * 16;
  v8f c[4];
#pragma unroll
  for (int mi = 0; mi < 4; ++mi)
#pragma unroll
    for (int r = 0; r < 8; ++r) c[mi][r] = 0.f;
#pragma unroll
  for (int k0 = 0; k0 < 512; k0 += 32) {
    const half_t* Asrc = (k0 < RNN) ? Af : Ab;
    int kk = k0 & (RNN - 1);
    v16h b = load_b_fragT(W, 512, n0, k0);
#pragma unroll
    for (int mi = 0; mi < 4; ++mi) {
      v16h a = load_a_frag(Asrc, RNN, mi * 16, kk);
      c[mi] = __builtin_amdgcn_wmma_f32_16x16x32_f16(false, a, false, b, (short)0, c[mi], false, false);
    }
  }
  float bv = db[n0 + nl];
#pragma unroll
  for (int mi = 0; mi < 4; ++mi)
#pragma unroll
    for (int r = 0; r < 8; ++r) {
      int m = mi * 16 + r + 8 * hh;
      float v = c[mi][r] + bv;
      dec[m * DECD + n0 + nl] = v > 0.f ? v : 0.f;   // relu
    }
  __syncthreads();

  if (threadIdx.x < BSZ) {
    int b = threadIdx.x;
    const float* mw = A.meanw[si];
    const float* mb = A.meanb[si];
    float m0 = mb[0], m1 = mb[1];
    for (int j = 0; j < DECD; ++j) {
      float d = dec[b * DECD + j];
      m0 += d * mw[j];
      m1 += d * mw[DECD + j];
    }
    const float* tgt = A.gt + (((size_t)(s + t) * BSZ + b) * P_DIM + p) * 2;
    float d0 = m0 - tgt[0];
    float d1 = m1 - tgt[1];
    redl[b] = fabsf(d0) + fabsf(d1);
    float e0 = d0 * 108.f, e1 = d1 * 72.f;
    redp[b] = sqrtf(e0 * e0 + e1 * e1);
  }
  __syncthreads();
  if (threadIdx.x == 0) {
    float ls = 0.f, ps = 0.f;
    for (int b = 0; b < BSZ; ++b) { ls += redl[b]; ps += redp[b]; }
    atomicAdd(A.acc, ls);
    atomicAdd(A.acc + 1, ps);
  }
}

__global__ void finalize_kernel(const float* acc, float* out) {
  const float count = (float)(P_DIM * (199 + 197 + 193 + 185));   // 17028
  out[0] = acc[0] * (10.f / (128.f * count));
  out[1] = acc[1] * (1.f / (64.f * count));
}

// ---- prep kernels -------------------------------------------------------
__global__ void zero_f_kernel(float* p, int n) {
  int i = blockIdx.x * blockDim.x + threadIdx.x;
  if (i < n) p[i] = 0.f;
}
__global__ void zero_h_kernel(half_t* p, int n) {
  int i = blockIdx.x * blockDim.x + threadIdx.x;
  if (i < n) p[i] = (half_t)0.f;
}
__global__ void cvt_kernel(const float* s, half_t* d, int n) {
  int i = blockIdx.x * blockDim.x + threadIdx.x;
  if (i < n) d[i] = (half_t)s[i];
}

// ---- host ---------------------------------------------------------------
struct HostParams {
  const float *data, *gt;
  const float *g_wih[2], *g_whh[2], *g_bih[2], *g_bhh[2];
  const float *b_wih[2], *b_whh[2], *b_bih[2], *b_bhh[2];
  const float *dec_w[4], *dec_b[4], *mean_w[4], *mean_b[4];
};

static void map_inputs(void* const* d_in, const int* in_sizes, HostParams& hp) {
  auto F = [&](int i) { return (const float*)d_in[i]; };
  if (in_sizes[0] == T_DIM * BSZ * P_DIM * 3) {
    // setup_inputs() insertion order
    hp.data = F(0); hp.gt = F(1);
    int i = 2;
    for (int l = 0; l < 2; ++l) { hp.g_wih[l] = F(i++); hp.g_whh[l] = F(i++); hp.g_bih[l] = F(i++); hp.g_bhh[l] = F(i++); }
    for (int l = 0; l < 2; ++l) { hp.b_wih[l] = F(i++); hp.b_whh[l] = F(i++); hp.b_bih[l] = F(i++); hp.b_bhh[l] = F(i++); }
    for (int s = 0; s < 4; ++s) { hp.dec_w[s] = F(i++); hp.dec_b[s] = F(i++); hp.mean_w[s] = F(i++); hp.mean_b[s] = F(i++); }
  } else {
    // jax pytree sorted-key flatten order
    int i = 0;
    for (int l = 0; l < 2; ++l) { hp.b_bhh[l] = F(i++); hp.b_bih[l] = F(i++); hp.b_whh[l] = F(i++); hp.b_wih[l] = F(i++); }
    hp.data = F(i++);
    for (int s = 0; s < 4; ++s) { hp.dec_b[s] = F(i++); hp.dec_w[s] = F(i++); hp.mean_b[s] = F(i++); hp.mean_w[s] = F(i++); }
    hp.gt = F(i++);
    for (int l = 0; l < 2; ++l) { hp.g_bhh[l] = F(i++); hp.g_bih[l] = F(i++); hp.g_whh[l] = F(i++); hp.g_wih[l] = F(i++); }
  }
}

extern "C" void kernel_launch(void* const* d_in, const int* in_sizes, int n_in,
                              void* d_out, int out_size, void* d_ws, size_t ws_size,
                              hipStream_t stream) {
  (void)n_in; (void)out_size; (void)ws_size;
  HostParams hp;
  map_inputs(d_in, in_sizes, hp);

  // workspace layout
  float* acc = (float*)d_ws;
  half_t* wh = (half_t*)((char*)d_ws + 256);
  const int WHH_N = G3 * RNN;    // 196608
  const int DECW_N = DECD * 512; // 65536
  half_t* fwhh0 = wh;
  half_t* fwhh1 = fwhh0 + WHH_N;
  half_t* fwih1 = fwhh1 + WHH_N;
  half_t* bwhh0 = fwih1 + WHH_N;
  half_t* bwhh1 = bwhh0 + WHH_N;
  half_t* bwih1 = bwhh1 + WHH_N;
  half_t* decw0 = bwih1 + WHH_N;
  half_t* hf = decw0 + 4 * (size_t)DECW_N;
  half_t* hb = hf + (size_t)NFWD * STEP_H;

  // zero accumulators and hb_trace[0] (initial hidden of the backward scan)
  zero_f_kernel<<<1, 64, 0, stream>>>(acc, 16);
  zero_h_kernel<<<(STEP_H + 255) / 256, 256, 0, stream>>>(hb, STEP_H);

  auto CV = [&](const float* s, half_t* d, int n) {
    cvt_kernel<<<(n + 255) / 256, 256, 0, stream>>>(s, d, n);
  };
  CV(hp.g_whh[0], fwhh0, WHH_N);
  CV(hp.g_whh[1], fwhh1, WHH_N);
  CV(hp.g_wih[1], fwih1, WHH_N);
  CV(hp.b_whh[0], bwhh0, WHH_N);
  CV(hp.b_whh[1], bwhh1, WHH_N);
  CV(hp.b_wih[1], bwih1, WHH_N);
  for (int s = 0; s < 4; ++s) CV(hp.dec_w[s], decw0 + (size_t)s * DECW_N, DECW_N);

  // persistent sequential scans (256 KB dynamic LDS)
  hipFuncSetAttribute((const void*)scan_kernel,
                      hipFuncAttributeMaxDynamicSharedMemorySize, 256 * 1024);
  ScanArgs sa;
  sa.data = hp.data; sa.gt = hp.gt;
  sa.f_wih0 = hp.g_wih[0]; sa.f_bih0 = hp.g_bih[0]; sa.f_bhh0 = hp.g_bhh[0];
  sa.f_bih1 = hp.g_bih[1]; sa.f_bhh1 = hp.g_bhh[1];
  sa.f_whh0 = fwhh0; sa.f_whh1 = fwhh1; sa.f_wih1 = fwih1;
  sa.b_wih0 = hp.b_wih[0]; sa.b_bih0 = hp.b_bih[0]; sa.b_bhh0 = hp.b_bhh[0];
  sa.b_bih1 = hp.b_bih[1]; sa.b_bhh1 = hp.b_bhh[1];
  sa.bw_whh0 = bwhh0; sa.bw_whh1 = bwhh1; sa.bw_wih1 = bwih1;
  sa.hf_trace = hf; sa.hb_trace = hb;
  scan_kernel<<<2, 512, 256 * 1024, stream>>>(sa);

  // decoder over all (s, p, t) tasks
  DecArgs da;
  da.gt = hp.gt; da.hf = hf; da.hb = hb; da.acc = acc;
  for (int s = 0; s < 4; ++s) {
    da.decw[s] = decw0 + (size_t)s * DECW_N;
    da.decb[s] = hp.dec_b[s];
    da.meanw[s] = hp.mean_w[s];
    da.meanb[s] = hp.mean_b[s];
  }
  const int ntasks = P_DIM * (199 + 197 + 193 + 185); // 17028
  dec_kernel<<<ntasks, 256, 0, stream>>>(da);

  finalize_kernel<<<1, 1, 0, stream>>>(acc, (float*)d_out);
}
